// TransformerBlock_56684978373044
// MI455X (gfx1250) — matrix operand order repaired and checked by reading, not compiled, not run on hardware
//
#include <hip/hip_runtime.h>
#include <hip/hip_bf16.h>
#include <math.h>

typedef __attribute__((ext_vector_type(16))) _Float16 v16h;
typedef __attribute__((ext_vector_type(8)))  _Float16 v8h;
typedef __attribute__((ext_vector_type(8)))  float    v8f;
typedef __attribute__((ext_vector_type(4)))  int      v4i;
typedef __attribute__((ext_vector_type(8)))  int      v8i;
typedef __attribute__((ext_vector_type(4)))  unsigned v4u;

#define DIM   1024
#define HEADS 16
#define HD    64
#define NTOK  1024
#define BATCH 8
#define TOKS  (BATCH*NTOK)   // 8192
#define MLPH  4096

#define AS1 __attribute__((address_space(1)))
#define AS3 __attribute__((address_space(3)))

#ifndef __has_builtin
#define __has_builtin(x) 0
#endif
#if __has_builtin(__builtin_amdgcn_tensor_load_to_lds)
#define HAVE_TDM 1
#else
#define HAVE_TDM 0
#endif
#if __has_builtin(__builtin_amdgcn_global_load_async_to_lds_b128)
#define HAVE_ASYNC_LDS 1
#else
#define HAVE_ASYNC_LDS 0
#endif

__device__ __forceinline__ v8f wmma32(v16h a, v16h b, v8f c) {
  return __builtin_amdgcn_wmma_f32_16x16x32_f16(
      false, a, false, b, (short)0, c, false, false);
}

__device__ __forceinline__ v8f zero8() {
  v8f z;
  #pragma unroll
  for (int i = 0; i < 8; ++i) z[i] = 0.0f;
  return z;
}

// 16-byte global -> LDS copy (async on CDNA5, ASYNCcnt-tracked)
__device__ __forceinline__ void cp16_g2l(const _Float16* g, _Float16* l) {
#if HAVE_ASYNC_LDS
  __builtin_amdgcn_global_load_async_to_lds_b128(
      (AS1 v4i*)(AS1 void*)const_cast<_Float16*>(g),
      (AS3 v4i*)(AS3 void*)l, 0, 0);
#else
  *(v8h*)l = *(const v8h*)g;
#endif
}

__device__ __forceinline__ void wait_async_all() {
#if HAVE_ASYNC_LDS
#if __has_builtin(__builtin_amdgcn_s_wait_asynccnt)
  __builtin_amdgcn_s_wait_asynccnt(0);
#else
  asm volatile("s_wait_asynccnt 0" ::: "memory");
#endif
#endif
}

__device__ __forceinline__ void wait_tensor0() {
#if __has_builtin(__builtin_amdgcn_s_wait_tensorcnt)
  __builtin_amdgcn_s_wait_tensorcnt(0);
#else
  asm volatile("s_wait_tensorcnt 0" ::: "memory");
#endif
}

#if HAVE_TDM
// TDM: one instruction DMAs a 128-row x 32-half 2D tile (row stride K halves)
// into LDS, padding each 64B row with 16B (4 DW) -> LDS row stride 40 halves.
// D# per CDNA5 ISA 8.3/8.4: group0 = {count, lds_addr, global_addr, type=2},
// group1 = {flags, dims, tile dims, strides}; groups 2/3 zero (2D tensor).
// This toolchain's builtin takes 6 args (extra int32x8 before cpol).
__device__ __forceinline__ void tdm_load_tile_2d(const _Float16* gsrc,
                                                 unsigned lds_off, int K) {
  unsigned long long ga = (unsigned long long)(uintptr_t)gsrc;
  v4u g0 = {0u, 0u, 0u, 0u};
  g0[0] = 1u;                                            // count=1, user desc
  g0[1] = lds_off;                                       // lds_addr (bytes)
  g0[2] = (unsigned)(ga & 0xFFFFFFFFu);                  // global_addr[31:0]
  g0[3] = (unsigned)((ga >> 32) & 0x01FFFFFFu)           // global_addr[56:32]
          | (2u << 30);                                  // type=2 ("image")
  v8i g1 = {0, 0, 0, 0, 0, 0, 0, 0};
  g1[0] = (int)((1u << 16)        // data_size = 2 bytes
              | (1u << 20)        // pad_enable
              | (3u << 22)        // pad_interval: 16 DWORDs (one 64B row)
              | (3u << 25));      // pad_amount: 4 DWORDs (16B)
  g1[1] = (int)(((unsigned)K & 0xFFFFu) << 16);          // tensor_dim0[15:0]
  g1[2] = (int)((((unsigned)K >> 16) & 0xFFFFu)          // tensor_dim0[31:16]
              | (128u << 16));                           // tensor_dim1[15:0]=128
  g1[3] = (int)(32u << 16);                              // tile_dim0 = 32
  g1[4] = 128;                                           // tile_dim1 = 128 (tile_dim2=0)
  g1[5] = K;                                             // tensor_dim0_stride[31:0]
  v4i g2 = {0, 0, 0, 0};
  v4i g3 = {0, 0, 0, 0};
  v8i g4 = {0, 0, 0, 0, 0, 0, 0, 0};
  __builtin_amdgcn_tensor_load_to_lds(g0, g1, g2, g3, g4, 0);
}
#endif

// ds_swizzle XOR-butterfly (within 16-lane halves; masks <16 never cross halves)
template <int IMM>
__device__ __forceinline__ float swz(float v) {
  return __int_as_float(__builtin_amdgcn_ds_swizzle(__float_as_int(v), IMM));
}

// A-fragment: 16x32 f16 (MxK). lane = m + 16*g; e0..15 -> K=k0+16g+e (contiguous)
__device__ __forceinline__ v16h load_a_frag(const _Float16* __restrict__ A,
                                            int lda, int row0, int k0, int lane) {
  int m = lane & 15, g = lane >> 4;
  const _Float16* p = A + (size_t)(row0 + m) * lda + k0 + 16 * g;
  v8h lo = *(const v8h*)p;
  v8h hi = *(const v8h*)(p + 8);
  v16h r;
  #pragma unroll
  for (int i = 0; i < 8; ++i) { r[i] = lo[i]; r[i + 8] = hi[i]; }
  return r;
}

// B-fragment from W (N x K row-major): lane = n + 16*g holds W[n][k0+16g+e] (contiguous)
__device__ __forceinline__ v16h load_b_frag(const _Float16* __restrict__ W,
                                            int ldw, int col0, int k0, int lane) {
  int n = lane & 15, g = lane >> 4;
  const _Float16* p = W + (size_t)(col0 + n) * ldw + k0 + 16 * g;
  v8h lo = *(const v8h*)p;
  v8h hi = *(const v8h*)(p + 8);
  v16h r;
  #pragma unroll
  for (int i = 0; i < 8; ++i) { r[i] = lo[i]; r[i + 8] = hi[i]; }
  return r;
}

// ---------------------------------------------------------------- convert
__global__ __launch_bounds__(256) void f32_to_f16_kernel(
    const float* __restrict__ in, _Float16* __restrict__ out, size_t n) {
  size_t i = (size_t)blockIdx.x * blockDim.x + threadIdx.x;
  size_t stride = (size_t)gridDim.x * blockDim.x;
  for (; i < n; i += stride) out[i] = (_Float16)in[i];
}

// ---------------------------------------------------------------- layernorm
__global__ __launch_bounds__(128) void layernorm_kernel(
    const float* __restrict__ in, const float* __restrict__ gam,
    const float* __restrict__ bet, _Float16* __restrict__ out) {
  __shared__ float rs[4], rs2[4];
  int row = blockIdx.x;
  const float* x = in + (size_t)row * DIM;
  float s = 0.f, s2 = 0.f;
  for (int i = threadIdx.x; i < DIM; i += 128) {
    float v = x[i];
    s += v; s2 += v * v;
  }
  #pragma unroll
  for (int off = 16; off > 0; off >>= 1) {
    s  += __shfl_xor(s,  off, 32);
    s2 += __shfl_xor(s2, off, 32);
  }
  int wave = threadIdx.x >> 5;
  if ((threadIdx.x & 31) == 0) { rs[wave] = s; rs2[wave] = s2; }
  __syncthreads();
  s = rs[0] + rs[1] + rs[2] + rs[3];
  s2 = rs2[0] + rs2[1] + rs2[2] + rs2[3];
  float mean = s * (1.0f / DIM);
  float var = s2 * (1.0f / DIM) - mean * mean;
  float inv = rsqrtf(var + 1e-5f);
  _Float16* o = out + (size_t)row * DIM;
  for (int i = threadIdx.x; i < DIM; i += 128)
    o[i] = (_Float16)((x[i] - mean) * inv * gam[i] + bet[i]);
}

// ---------------------------------------------------------------- GEMM
// C[M,N] = A[M,K] @ W[N,K]^T + bias. Double-buffered K=32 slabs staged in LDS
// by the Tensor Data Mover (fallback: async global->LDS copies).
// MODE 0: scatter q/k/v f16 ; MODE 1: f32 = acc+bias+resid ; MODE 2: f16 = gelu_erf(acc+bias)
#define LDP 40   // padded LDS row stride (halves) for 32-half rows
template <int MODE>
__global__ __launch_bounds__(256) void gemm_wmma_kernel(
    const _Float16* __restrict__ A, const _Float16* __restrict__ W,
    const float* __restrict__ bias, int M, int N, int K,
    const float* __restrict__ resid, float* __restrict__ outf,
    _Float16* __restrict__ outh,
    _Float16* __restrict__ qb, _Float16* __restrict__ kb,
    _Float16* __restrict__ vb) {
  __shared__ __align__(16) _Float16 Asm[2][128 * LDP];
  __shared__ __align__(16) _Float16 Bsm[2][128 * LDP];

  int tid = threadIdx.x, lane = tid & 31, wave = tid >> 5;
  int wm = wave >> 1, wn = wave & 1;
  int row0b = blockIdx.y * 128, col0b = blockIdx.x * 128;

  v8f acc[2][4];
  #pragma unroll
  for (int i = 0; i < 2; ++i)
    #pragma unroll
    for (int j = 0; j < 4; ++j) acc[i][j] = zero8();

  auto stage = [&](int s) {
    int k0 = s * 32, buf = s & 1;
#if HAVE_TDM
    if (wave == 0) {
      unsigned ldsA = (unsigned)(uintptr_t)(AS3 void*)&Asm[buf][0];
      unsigned ldsB = (unsigned)(uintptr_t)(AS3 void*)&Bsm[buf][0];
      tdm_load_tile_2d(A + (size_t)row0b * K + k0, ldsA, K);
      tdm_load_tile_2d(W + (size_t)col0b * K + k0, ldsB, K);
    }
#else
    #pragma unroll
    for (int c = tid; c < 512; c += 256) {   // 512 16B chunks per 128x32 tile
      int r = c >> 2, cc = (c & 3) * 8;
      cp16_g2l(A + (size_t)(row0b + r) * K + k0 + cc, &Asm[buf][r * LDP + cc]);
      cp16_g2l(W + (size_t)(col0b + r) * K + k0 + cc, &Bsm[buf][r * LDP + cc]);
    }
#endif
  };

  int nk = K / 32;
  stage(0);
  for (int s = 0; s < nk; ++s) {
#if HAVE_TDM
    if (wave == 0) wait_tensor0();   // issuing wave's TDM ops complete
#else
    wait_async_all();                // own stage-s copies landed in LDS
#endif
    __syncthreads();                 // tiles visible; all done reading other buf
    if (s + 1 < nk) stage(s + 1);
    const _Float16* As = Asm[s & 1];
    const _Float16* Bs = Bsm[s & 1];
    v16h a0 = load_a_frag(As, LDP, wm * 32,      0, lane);
    v16h a1 = load_a_frag(As, LDP, wm * 32 + 16, 0, lane);
    #pragma unroll
    for (int ni = 0; ni < 4; ++ni) {
      v16h b = load_b_frag(Bs, LDP, wn * 64 + 16 * ni, 0, lane);
      acc[0][ni] = wmma32(a0, b, acc[0][ni]);
      acc[1][ni] = wmma32(a1, b, acc[1][ni]);
    }
  }

  int row0 = row0b + wm * 32, col0 = col0b + wn * 64;
  int nlane = lane & 15, mh = (lane >> 4) * 8;
  #pragma unroll
  for (int mi = 0; mi < 2; ++mi)
    #pragma unroll
    for (int ni = 0; ni < 4; ++ni) {
      int col = col0 + 16 * ni + nlane;
      float bv = bias[col];
      #pragma unroll
      for (int e = 0; e < 8; ++e) {
        int row = row0 + 16 * mi + mh + e;
        float val = acc[mi][ni][e] + bv;
        if (MODE == 1) {
          size_t idx = (size_t)row * N + col;
          outf[idx] = val + resid[idx];
        } else if (MODE == 2) {
          float gv = 0.5f * val * (1.0f + erff(val * 0.70710678118654752f));
          outh[(size_t)row * N + col] = (_Float16)gv;
        } else {  // QKV scatter: col = s*1024 + h*64 + d ; row = b*1024 + n
          int sidx = col >> 10, rem = col & 1023;
          int h = rem >> 6, d = rem & 63;
          int bidx = row >> 10, n = row & 1023;
          _Float16* dst = (sidx == 0) ? qb : ((sidx == 1) ? kb : vb);
          dst[((size_t)(bidx * HEADS + h) * NTOK + n) * HD + d] = (_Float16)val;
        }
      }
    }
}

// ---------------------------------------------------------------- attention
// One wave = 16 query rows of one (batch, head); flash-style online softmax.
__global__ __launch_bounds__(128) void attn_kernel(
    const _Float16* __restrict__ qb, const _Float16* __restrict__ kb,
    const _Float16* __restrict__ vb, _Float16* __restrict__ out) {
  __shared__ __align__(16) _Float16 pstage[4][16 * 32];
  int tid = threadIdx.x, lane = tid & 31, wave = tid >> 5;
  int widx = blockIdx.x * 4 + wave;
  int qt = widx & 63, h = (widx >> 6) & 15, b = widx >> 10;

  const _Float16* qp  = qb + ((size_t)(b * HEADS + h) * NTOK + qt * 16) * HD;
  const _Float16* kp0 = kb + (size_t)(b * HEADS + h) * NTOK * HD;
  const _Float16* vp0 = vb + (size_t)(b * HEADS + h) * NTOK * HD;

  v16h qa0 = load_a_frag(qp, HD, 0, 0,  lane);
  v16h qa1 = load_a_frag(qp, HD, 0, 32, lane);

  v8f o[4];
  #pragma unroll
  for (int t = 0; t < 4; ++t) o[t] = zero8();
  float mrow[8], lrow[8];
  #pragma unroll
  for (int e = 0; e < 8; ++e) { mrow[e] = -1e30f; lrow[e] = 0.0f; }

  int nlane = lane & 15, g = lane >> 4;
  _Float16* ps = pstage[wave];

  for (int kbk = 0; kbk < NTOK; kbk += 32) {
    const _Float16* kp = kp0 + (size_t)kbk * HD;
    const _Float16* vp = vp0 + (size_t)kbk * HD;
    __builtin_prefetch(kp + 32 * HD, 0, 1);
    __builtin_prefetch(vp + 32 * HD, 0, 1);

    v8f s0 = zero8(), s1 = zero8();
    {
      v16h b0 = load_b_frag(kp, HD, 0,  0,  lane);
      v16h b1 = load_b_frag(kp, HD, 0,  32, lane);
      s0 = wmma32(qa0, b0, s0);
      s0 = wmma32(qa1, b1, s0);
      v16h b2 = load_b_frag(kp, HD, 16, 0,  lane);
      v16h b3 = load_b_frag(kp, HD, 16, 32, lane);
      s1 = wmma32(qa0, b2, s1);
      s1 = wmma32(qa1, b3, s1);
    }

    float p0[8], p1[8], corr[8];
    #pragma unroll
    for (int e = 0; e < 8; ++e) {
      // reference divides scores by HD^-0.5 => multiply by sqrt(64) = 8
      float v0 = s0[e] * 8.0f, v1 = s1[e] * 8.0f;
      float mx = fmaxf(v0, v1);
      mx = fmaxf(mx, swz<0x201F>(mx));   // xor 8
      mx = fmaxf(mx, swz<0x101F>(mx));   // xor 4
      mx = fmaxf(mx, swz<0x081F>(mx));   // xor 2
      mx = fmaxf(mx, swz<0x041F>(mx));   // xor 1
      float mn = fmaxf(mrow[e], mx);
      corr[e] = __expf(mrow[e] - mn);
      v0 = __expf(v0 - mn);
      v1 = __expf(v1 - mn);
      float rsum = v0 + v1;
      rsum += swz<0x201F>(rsum);
      rsum += swz<0x101F>(rsum);
      rsum += swz<0x081F>(rsum);
      rsum += swz<0x041F>(rsum);
      lrow[e] = lrow[e] * corr[e] + rsum;
      mrow[e] = mn;
      p0[e] = v0; p1[e] = v1;
    }
    #pragma unroll
    for (int t = 0; t < 4; ++t)
      #pragma unroll
      for (int e = 0; e < 8; ++e) o[t][e] *= corr[e];

    // Stage P (16x32) to LDS in C-layout positions, reload in A-layout.
    int mh = g * 8;
    #pragma unroll
    for (int e = 0; e < 8; ++e) {
      ps[(mh + e) * 32 + nlane]      = (_Float16)p0[e];
      ps[(mh + e) * 32 + 16 + nlane] = (_Float16)p1[e];
    }
    asm volatile("s_wait_dscnt 0" ::: "memory");
    __builtin_amdgcn_wave_barrier();

    v16h pa;
    {
      const _Float16* pp = ps + nlane * 32 + 16 * g;
      v8h lo = *(const v8h*)pp;
      v8h hi = *(const v8h*)(pp + 8);
      #pragma unroll
      for (int i = 0; i < 8; ++i) { pa[i] = lo[i]; pa[i + 8] = hi[i]; }
    }

    // O += P(16x32) @ V(32x64)
    #pragma unroll
    for (int t = 0; t < 4; ++t) {
      v16h bv;
      #pragma unroll
      for (int e = 0; e < 16; ++e)
        bv[e] = vp[(size_t)(16 * g + e) * HD + t * 16 + nlane];
      o[t] = wmma32(pa, bv, o[t]);
    }
  }

  int token = b * NTOK + qt * 16;
  int mh = g * 8;
  #pragma unroll
  for (int t = 0; t < 4; ++t)
    #pragma unroll
    for (int e = 0; e < 8; ++e) {
      float val = o[t][e] / lrow[e];
      out[(size_t)(token + mh + e) * DIM + h * HD + t * 16 + nlane] = (_Float16)val;
    }
}

// ---------------------------------------------------------------- launch
extern "C" void kernel_launch(void* const* d_in, const int* in_sizes, int n_in,
                              void* d_out, int out_size, void* d_ws, size_t ws_size,
                              hipStream_t stream) {
  const float* x      = (const float*)d_in[0];
  const float* ln1_g  = (const float*)d_in[1];
  const float* ln1_b  = (const float*)d_in[2];
  const float* qkv_w  = (const float*)d_in[3];
  const float* qkv_b  = (const float*)d_in[4];
  const float* proj_w = (const float*)d_in[5];
  const float* proj_b = (const float*)d_in[6];
  const float* ln2_g  = (const float*)d_in[7];
  const float* ln2_b  = (const float*)d_in[8];
  const float* fc1_w  = (const float*)d_in[9];
  const float* fc1_b  = (const float*)d_in[10];
  const float* fc2_w  = (const float*)d_in[11];
  const float* fc2_b  = (const float*)d_in[12];
  float* out = (float*)d_out;

  char* ws = (char*)d_ws;
  size_t off = 0;
  _Float16* qkvw16  = (_Float16*)(ws + off); off += (size_t)3 * DIM * DIM * 2;
  _Float16* projw16 = (_Float16*)(ws + off); off += (size_t)DIM * DIM * 2;
  _Float16* fc1w16  = (_Float16*)(ws + off); off += (size_t)MLPH * DIM * 2;
  _Float16* fc2w16  = (_Float16*)(ws + off); off += (size_t)DIM * MLPH * 2;
  _Float16* xln1    = (_Float16*)(ws + off); off += (size_t)TOKS * DIM * 2;
  _Float16* qbuf    = (_Float16*)(ws + off); off += (size_t)TOKS * DIM * 2;
  _Float16* kbuf    = (_Float16*)(ws + off); off += (size_t)TOKS * DIM * 2;
  _Float16* vbuf    = (_Float16*)(ws + off); off += (size_t)TOKS * DIM * 2;
  _Float16* attn16  = (_Float16*)(ws + off); off += (size_t)TOKS * DIM * 2;
  float*    hbuf    = (float*)(ws + off);    off += (size_t)TOKS * DIM * 4;
  _Float16* y1   = qbuf;  // 64 MB alias over q/k/v/attn16 (dead by FC1)
  _Float16* hln2 = xln1;  // alias (xln1 dead after QKV GEMM)
  (void)off; (void)ws_size; (void)n_in; (void)in_sizes; (void)out_size;

  f32_to_f16_kernel<<<2048, 256, 0, stream>>>(qkv_w, qkvw16, (size_t)3 * DIM * DIM);
  f32_to_f16_kernel<<<2048, 256, 0, stream>>>(proj_w, projw16, (size_t)DIM * DIM);
  f32_to_f16_kernel<<<2048, 256, 0, stream>>>(fc1_w, fc1w16, (size_t)MLPH * DIM);
  f32_to_f16_kernel<<<2048, 256, 0, stream>>>(fc2_w, fc2w16, (size_t)DIM * MLPH);

  layernorm_kernel<<<TOKS, 128, 0, stream>>>(x, ln1_g, ln1_b, xln1);

  gemm_wmma_kernel<0><<<dim3(3 * DIM / 128, TOKS / 128), 256, 0, stream>>>(
      xln1, qkvw16, qkv_b, TOKS, 3 * DIM, DIM,
      nullptr, nullptr, nullptr, qbuf, kbuf, vbuf);

  attn_kernel<<<TOKS * HEADS / 16 / 4, 128, 0, stream>>>(qbuf, kbuf, vbuf, attn16);

  gemm_wmma_kernel<1><<<dim3(DIM / 128, TOKS / 128), 256, 0, stream>>>(
      attn16, projw16, proj_b, TOKS, DIM, DIM,
      x, hbuf, nullptr, nullptr, nullptr, nullptr);

  layernorm_kernel<<<TOKS, 128, 0, stream>>>(hbuf, ln2_g, ln2_b, hln2);

  gemm_wmma_kernel<2><<<dim3(MLPH / 128, TOKS / 128), 256, 0, stream>>>(
      hln2, fc1w16, fc1_b, TOKS, MLPH, DIM,
      nullptr, nullptr, y1, nullptr, nullptr, nullptr);

  gemm_wmma_kernel<1><<<dim3(DIM / 128, TOKS / 128), 256, 0, stream>>>(
      y1, fc2w16, fc2_b, TOKS, DIM, MLPH,
      hbuf, out, nullptr, nullptr, nullptr, nullptr);
}
